// Model_78288663872083
// MI455X (gfx1250) — matrix operand order repaired and checked by reading, not compiled, not run on hardware
//
#include <hip/hip_runtime.h>

#define L 128
#define WPB 2   // 16-row wave tiles per block (64 threads)

typedef _Float16 v16h __attribute__((ext_vector_type(16)));
typedef _Float16 v8h  __attribute__((ext_vector_type(8)));
typedef float    v8f  __attribute__((ext_vector_type(8)));

__device__ inline void lds_fence() { asm volatile("s_wait_dscnt 0" ::: "memory"); }

// ---- A fragment from LDS f16 row-major tile (ISA 7.12.2 16-bit A 16x32) ----
__device__ inline v16h load_afrag(const _Float16* X, int xs, int lane, int kt) {
  const int m  = lane & 15;
  const int kb = ((lane >> 4) << 3) + (kt << 5);
  const _Float16* base = X + m * xs + kb;
  v8h lo = *(const v8h*)(base);
  v8h hi = *(const v8h*)(base + 16);
  v16h a;
#pragma unroll
  for (int i = 0; i < 8; ++i) { a[i] = lo[i]; a[i + 8] = hi[i]; }
  return a;
}

// ---- 16xK @ KxL layer, bias + relu, f16 result to LDS ----
__device__ inline void wave_layer_relu(const _Float16* __restrict__ Wsw,
                                       const float* __restrict__ bias,
                                       const _Float16* X, int xs, int KT, int lane,
                                       _Float16* H, int hs) {
  const int n_lo = lane & 15;
  const int m_hi = (lane >> 4) << 3;
  for (int nt = 0; nt < 8; ++nt) {
    float bv = bias[nt * 16 + n_lo];
    v8f c;
#pragma unroll
    for (int i = 0; i < 8; ++i) c[i] = bv;
    const _Float16* wp = Wsw + ((size_t)(nt * KT) * 32 + lane) * 16;
    for (int kt = 0; kt < KT; ++kt) {
      v16h a = load_afrag(X, xs, lane, kt);
      v16h b = *(const v16h*)(wp + (size_t)kt * 512);
      c = __builtin_amdgcn_wmma_f32_16x16x32_f16(false, a, false, b, (short)0, c, false, false);
    }
#pragma unroll
    for (int j = 0; j < 8; ++j) {
      float v = c[j];
      v = v > 0.f ? v : 0.f;
      H[(m_hi + j) * hs + nt * 16 + n_lo] = (_Float16)v;
    }
  }
}

// ---- final linear layer (no relu), fp32 result to LDS (stride 128) ----
__device__ inline void wave_layer_out(const _Float16* __restrict__ Wsw,
                                      const float* __restrict__ bias,
                                      const _Float16* X, int xs, int KT, int lane,
                                      float* OUT) {
  const int n_lo = lane & 15;
  const int m_hi = (lane >> 4) << 3;
  for (int nt = 0; nt < 8; ++nt) {
    float bv = bias[nt * 16 + n_lo];
    v8f c;
#pragma unroll
    for (int i = 0; i < 8; ++i) c[i] = bv;
    const _Float16* wp = Wsw + ((size_t)(nt * KT) * 32 + lane) * 16;
    for (int kt = 0; kt < KT; ++kt) {
      v16h a = load_afrag(X, xs, lane, kt);
      v16h b = *(const v16h*)(wp + (size_t)kt * 512);
      c = __builtin_amdgcn_wmma_f32_16x16x32_f16(false, a, false, b, (short)0, c, false, false);
    }
#pragma unroll
    for (int j = 0; j < 8; ++j)
      OUT[(m_hi + j) * L + nt * 16 + n_lo] = c[j];
  }
}

__device__ inline void store4h(_Float16* p, float4 v) {
  p[0] = (_Float16)v.x; p[1] = (_Float16)v.y; p[2] = (_Float16)v.z; p[3] = (_Float16)v.w;
}

// ---- weight swizzle: fp32 [K][128] -> f16 fragment order [nt][kt][lane][h] ----
// B layout (ISA 7.12.4): lane&15 = column N, K = 8*(lane>>4) + (h&7) + 16*(h>>3)
__global__ void swizzle_w_kernel(const float* __restrict__ src, _Float16* __restrict__ dst,
                                 int K, int Kpad) {
  int t = blockIdx.x * blockDim.x + threadIdx.x;
  int total = Kpad * 128;
  if (t >= total) return;
  int h = t & 15;
  int lane = (t >> 4) & 31;
  int tileIdx = t >> 9;
  int KT = Kpad >> 5;
  int kt = tileIdx % KT;
  int nt = tileIdx / KT;
  int k = kt * 32 + ((lane >> 4) << 3) + (h & 7) + ((h >> 3) << 4);
  int n = nt * 16 + (lane & 15);
  float v = (k < K) ? src[(size_t)k * 128 + n] : 0.f;
  dst[t] = (_Float16)v;
}

__global__ void zero_kernel(float* __restrict__ p, size_t n) {
  size_t i = blockIdx.x * (size_t)blockDim.x + threadIdx.x;
  if (i < n) p[i] = 0.f;
}

// ================= encoder: nodes =================
__global__ __launch_bounds__(64) void encode_nodes_kernel(
    const float* __restrict__ wp, const float* __restrict__ pwp,
    const int* __restrict__ ntype,
    const float* __restrict__ nmean, const float* __restrict__ nstd,
    const _Float16* __restrict__ w0s, const _Float16* __restrict__ w1s,
    const _Float16* __restrict__ w2s,
    const float* __restrict__ b0, const float* __restrict__ b1, const float* __restrict__ b2,
    const float* __restrict__ g, const float* __restrict__ bt,
    float* __restrict__ node_lat, int Nn) {
  __shared__ __align__(32) _Float16 sX[WPB][16 * 40];
  __shared__ __align__(32) _Float16 sH1[WPB][16 * 136];
  __shared__ __align__(32) _Float16 sH2[WPB][16 * 136];
  __shared__ __align__(32) float    sO[WPB][16 * L];
  int wave = threadIdx.x >> 5, lane = threadIdx.x & 31;
  int tile = blockIdx.x * WPB + wave;
  if (tile >= (Nn >> 4)) return;
  _Float16* X = sX[wave]; _Float16* H1 = sH1[wave]; _Float16* H2 = sH2[wave];
  float* OUT = sO[wave];
  int base = tile << 4;
  if (lane < 16) {
    int n = base + lane;
    int ty = ntype[n];
    _Float16* xr = X + lane * 40;
    for (int c = 0; c < 3; ++c) {
      float v = wp[(size_t)n * 3 + c] - pwp[(size_t)n * 3 + c];
      xr[c] = (_Float16)((v - nmean[c]) / nstd[c]);
    }
    for (int j = 0; j < 9; ++j)
      xr[3 + j] = (_Float16)(((ty == j ? 1.f : 0.f) - nmean[3 + j]) / nstd[3 + j]);
    for (int k = 12; k < 40; ++k) xr[k] = (_Float16)0.f;
  }
  lds_fence();
  wave_layer_relu(w0s, b0, X, 40, 1, lane, H1, 136);
  lds_fence();
  wave_layer_relu(w1s, b1, H1, 136, 4, lane, H2, 136);
  lds_fence();
  wave_layer_out(w2s, b2, H2, 136, 4, lane, OUT);
  lds_fence();
  if (lane < 16) {
    float sum = 0.f, sq = 0.f;
    for (int n = 0; n < L; ++n) { float v = OUT[lane * L + n]; sum += v; sq += v * v; }
    float mean = sum * (1.f / L);
    float var = sq * (1.f / L) - mean * mean;
    float rstd = rsqrtf(var + 1e-5f);
    for (int n = 0; n < L; ++n)
      OUT[lane * L + n] = (OUT[lane * L + n] - mean) * rstd * g[n] + bt[n];
  }
  lds_fence();
  for (int i = 0; i < 16; ++i)
    *(float4*)(node_lat + (size_t)(base + i) * L + lane * 4) =
        *(const float4*)(OUT + i * L + lane * 4);
}

// ================= encoder: edges =================
__global__ __launch_bounds__(64) void encode_edges_kernel(
    const float* __restrict__ wp, const float* __restrict__ mp,
    const int* __restrict__ senders, const int* __restrict__ receivers,
    const float* __restrict__ emean, const float* __restrict__ estd,
    const _Float16* __restrict__ w0s, const _Float16* __restrict__ w1s,
    const _Float16* __restrict__ w2s,
    const float* __restrict__ b0, const float* __restrict__ b1, const float* __restrict__ b2,
    const float* __restrict__ g, const float* __restrict__ bt,
    float* __restrict__ edge_lat, int E, int ntiles) {
  __shared__ __align__(32) _Float16 sX[WPB][16 * 40];
  __shared__ __align__(32) _Float16 sH1[WPB][16 * 136];
  __shared__ __align__(32) _Float16 sH2[WPB][16 * 136];
  __shared__ __align__(32) float    sO[WPB][16 * L];
  int wave = threadIdx.x >> 5, lane = threadIdx.x & 31;
  int tile = blockIdx.x * WPB + wave;
  if (tile >= ntiles) return;
  _Float16* X = sX[wave]; _Float16* H1 = sH1[wave]; _Float16* H2 = sH2[wave];
  float* OUT = sO[wave];
  int base = tile << 4;
  if (lane < 16) {
    int e = base + lane; if (e >= E) e = E - 1;
    int s = senders[e], r = receivers[e];
    float rw0 = wp[(size_t)s * 3 + 0] - wp[(size_t)r * 3 + 0];
    float rw1 = wp[(size_t)s * 3 + 1] - wp[(size_t)r * 3 + 1];
    float rw2 = wp[(size_t)s * 3 + 2] - wp[(size_t)r * 3 + 2];
    float nw = sqrtf(rw0 * rw0 + rw1 * rw1 + rw2 * rw2);
    float rm0 = mp[(size_t)s * 2 + 0] - mp[(size_t)r * 2 + 0];
    float rm1 = mp[(size_t)s * 2 + 1] - mp[(size_t)r * 2 + 1];
    float nm = sqrtf(rm0 * rm0 + rm1 * rm1);
    float f[7] = {rw0, rw1, rw2, nw, rm0, rm1, nm};
    _Float16* xr = X + lane * 40;
    for (int c = 0; c < 7; ++c) xr[c] = (_Float16)((f[c] - emean[c]) / estd[c]);
    for (int k = 7; k < 40; ++k) xr[k] = (_Float16)0.f;
  }
  lds_fence();
  wave_layer_relu(w0s, b0, X, 40, 1, lane, H1, 136);
  lds_fence();
  wave_layer_relu(w1s, b1, H1, 136, 4, lane, H2, 136);
  lds_fence();
  wave_layer_out(w2s, b2, H2, 136, 4, lane, OUT);
  lds_fence();
  if (lane < 16) {
    float sum = 0.f, sq = 0.f;
    for (int n = 0; n < L; ++n) { float v = OUT[lane * L + n]; sum += v; sq += v * v; }
    float mean = sum * (1.f / L);
    float var = sq * (1.f / L) - mean * mean;
    float rstd = rsqrtf(var + 1e-5f);
    for (int n = 0; n < L; ++n)
      OUT[lane * L + n] = (OUT[lane * L + n] - mean) * rstd * g[n] + bt[n];
  }
  lds_fence();
  for (int i = 0; i < 16; ++i) {
    int e = base + i;
    if (e < E)
      *(float4*)(edge_lat + (size_t)e * L + lane * 4) =
          *(const float4*)(OUT + i * L + lane * 4);
  }
}

// ================= MP step: edge update + fused segment_sum =================
__global__ __launch_bounds__(64) void edge_step_kernel(
    const int* __restrict__ senders, const int* __restrict__ receivers,
    float* __restrict__ edge_lat, const float* __restrict__ node_lat,
    float* __restrict__ agg,
    const _Float16* __restrict__ w0s, const _Float16* __restrict__ w1s,
    const _Float16* __restrict__ w2s,
    const float* __restrict__ b0, const float* __restrict__ b1, const float* __restrict__ b2,
    const float* __restrict__ g, const float* __restrict__ bt,
    int E, int ntiles) {
  __shared__ __align__(32) _Float16 sX[WPB][16 * 392];   // ein: [edge|sender|recv]
  __shared__ __align__(32) _Float16 sH1[WPB][16 * 136];
  __shared__ __align__(32) _Float16 sH2[WPB][16 * 136];
  int wave = threadIdx.x >> 5, lane = threadIdx.x & 31;
  int tile = blockIdx.x * WPB + wave;
  if (tile >= ntiles) return;
  _Float16* X = sX[wave]; _Float16* H1 = sH1[wave]; _Float16* H2 = sH2[wave];
  float* OUT = (float*)X;               // reuse X after layer0 (16*128 fp32 fits)
  int base = tile << 4;
  for (int i = 0; i < 16; ++i) {
    int e = base + i; if (e >= E) e = E - 1;
    int s = senders[e], r = receivers[e];
    float4 fe = *(const float4*)(edge_lat + (size_t)e * L + lane * 4);
    float4 fs = *(const float4*)(node_lat + (size_t)s * L + lane * 4);
    float4 fr = *(const float4*)(node_lat + (size_t)r * L + lane * 4);
    store4h(X + i * 392 + 0   + lane * 4, fe);
    store4h(X + i * 392 + 128 + lane * 4, fs);
    store4h(X + i * 392 + 256 + lane * 4, fr);
  }
  lds_fence();
  wave_layer_relu(w0s, b0, X, 392, 12, lane, H1, 136);
  lds_fence();
  wave_layer_relu(w1s, b1, H1, 136, 4, lane, H2, 136);
  lds_fence();
  wave_layer_out(w2s, b2, H2, 136, 4, lane, OUT);
  lds_fence();
  if (lane < 16) {
    float sum = 0.f, sq = 0.f;
    for (int n = 0; n < L; ++n) { float v = OUT[lane * L + n]; sum += v; sq += v * v; }
    float mean = sum * (1.f / L);
    float var = sq * (1.f / L) - mean * mean;
    float rstd = rsqrtf(var + 1e-5f);
    int e = base + lane;
    if (e < E)
      for (int n = 0; n < L; ++n)
        OUT[lane * L + n] = (OUT[lane * L + n] - mean) * rstd * g[n] + bt[n] +
                            edge_lat[(size_t)e * L + n];
  }
  lds_fence();
  for (int i = 0; i < 16; ++i) {
    int e = base + i;
    if (e < E)
      *(float4*)(edge_lat + (size_t)e * L + lane * 4) =
          *(const float4*)(OUT + i * L + lane * 4);
  }
  for (int i = 0; i < 16; ++i) {
    int e = base + i;
    if (e < E) {
      int r = receivers[e];
      for (int n = lane; n < L; n += 32)
        atomicAdd(agg + (size_t)r * L + n, OUT[i * L + n]);
    }
  }
}

// ================= MP step: node update (reads+zeroes agg) =================
__global__ __launch_bounds__(64) void node_step_kernel(
    float* __restrict__ node_lat, float* __restrict__ agg,
    const _Float16* __restrict__ w0s, const _Float16* __restrict__ w1s,
    const _Float16* __restrict__ w2s,
    const float* __restrict__ b0, const float* __restrict__ b1, const float* __restrict__ b2,
    const float* __restrict__ g, const float* __restrict__ bt, int Nn) {
  __shared__ __align__(32) _Float16 sX[WPB][16 * 264];   // [node|agg]
  __shared__ __align__(32) _Float16 sH1[WPB][16 * 136];
  __shared__ __align__(32) _Float16 sH2[WPB][16 * 136];
  int wave = threadIdx.x >> 5, lane = threadIdx.x & 31;
  int tile = blockIdx.x * WPB + wave;
  if (tile >= (Nn >> 4)) return;
  _Float16* X = sX[wave]; _Float16* H1 = sH1[wave]; _Float16* H2 = sH2[wave];
  float* OUT = (float*)X;
  int base = tile << 4;
  float4 z4 = make_float4(0.f, 0.f, 0.f, 0.f);
  for (int i = 0; i < 16; ++i) {
    int n = base + i;
    float4 fn = *(const float4*)(node_lat + (size_t)n * L + lane * 4);
    float4 fa = *(const float4*)(agg + (size_t)n * L + lane * 4);
    *(float4*)(agg + (size_t)n * L + lane * 4) = z4;   // zero for next step
    store4h(X + i * 264 + 0   + lane * 4, fn);
    store4h(X + i * 264 + 128 + lane * 4, fa);
  }
  lds_fence();
  wave_layer_relu(w0s, b0, X, 264, 8, lane, H1, 136);
  lds_fence();
  wave_layer_relu(w1s, b1, H1, 136, 4, lane, H2, 136);
  lds_fence();
  wave_layer_out(w2s, b2, H2, 136, 4, lane, OUT);
  lds_fence();
  if (lane < 16) {
    float sum = 0.f, sq = 0.f;
    for (int n = 0; n < L; ++n) { float v = OUT[lane * L + n]; sum += v; sq += v * v; }
    float mean = sum * (1.f / L);
    float var = sq * (1.f / L) - mean * mean;
    float rstd = rsqrtf(var + 1e-5f);
    int nn = base + lane;
    for (int n = 0; n < L; ++n)
      OUT[lane * L + n] = (OUT[lane * L + n] - mean) * rstd * g[n] + bt[n] +
                          node_lat[(size_t)nn * L + n];
  }
  lds_fence();
  for (int i = 0; i < 16; ++i)
    *(float4*)(node_lat + (size_t)(base + i) * L + lane * 4) =
        *(const float4*)(OUT + i * L + lane * 4);
}

// ================= decoder + integrate =================
__global__ __launch_bounds__(64) void decode_kernel(
    const float* __restrict__ node_lat,
    const _Float16* __restrict__ w0s, const _Float16* __restrict__ w1s,
    const float* __restrict__ b0, const float* __restrict__ b1,
    const float* __restrict__ dW2, const float* __restrict__ db2,
    const float* __restrict__ omean, const float* __restrict__ ostd,
    const float* __restrict__ wp, const float* __restrict__ pwp,
    float* __restrict__ out, int Nn) {
  __shared__ __align__(32) _Float16 sX[WPB][16 * 136];
  __shared__ __align__(32) _Float16 sH1[WPB][16 * 136];
  __shared__ __align__(32) _Float16 sH2[WPB][16 * 136];
  int wave = threadIdx.x >> 5, lane = threadIdx.x & 31;
  int tile = blockIdx.x * WPB + wave;
  if (tile >= (Nn >> 4)) return;
  _Float16* X = sX[wave]; _Float16* H1 = sH1[wave]; _Float16* H2 = sH2[wave];
  int base = tile << 4;
  for (int i = 0; i < 16; ++i) {
    float4 fn = *(const float4*)(node_lat + (size_t)(base + i) * L + lane * 4);
    store4h(X + i * 136 + lane * 4, fn);
  }
  lds_fence();
  wave_layer_relu(w0s, b0, X, 136, 4, lane, H1, 136);
  lds_fence();
  wave_layer_relu(w1s, b1, H1, 136, 4, lane, H2, 136);
  lds_fence();
  if (lane < 16) {
    int n = base + lane;
    float a0 = db2[0], a1 = db2[1], a2 = db2[2];
    for (int k = 0; k < L; ++k) {
      float h = (float)H2[lane * 136 + k];
      a0 += h * dW2[k * 3 + 0];
      a1 += h * dW2[k * 3 + 1];
      a2 += h * dW2[k * 3 + 2];
    }
    a0 = a0 * ostd[0] + omean[0];
    a1 = a1 * ostd[1] + omean[1];
    a2 = a2 * ostd[2] + omean[2];
    out[(size_t)n * 3 + 0] = 2.f * wp[(size_t)n * 3 + 0] + a0 - pwp[(size_t)n * 3 + 0];
    out[(size_t)n * 3 + 1] = 2.f * wp[(size_t)n * 3 + 1] + a1 - pwp[(size_t)n * 3 + 1];
    out[(size_t)n * 3 + 2] = 2.f * wp[(size_t)n * 3 + 2] + a2 - pwp[(size_t)n * 3 + 2];
  }
}

// ======================= host launcher =======================
extern "C" void kernel_launch(void* const* d_in, const int* in_sizes, int n_in,
                              void* d_out, int out_size, void* d_ws, size_t ws_size,
                              hipStream_t stream) {
  const float* world_pos = (const float*)d_in[0];
  const float* prev_world_pos = (const float*)d_in[1];
  const float* mesh_pos = (const float*)d_in[2];
  const float* node_mean = (const float*)d_in[3];
  const float* node_std = (const float*)d_in[4];
  const float* edge_mean = (const float*)d_in[5];
  const float* edge_std = (const float*)d_in[6];
  const float* out_mean = (const float*)d_in[7];
  const float* out_std = (const float*)d_in[8];
  const float* enW0 = (const float*)d_in[9];  const float* enb0 = (const float*)d_in[10];
  const float* enW1 = (const float*)d_in[11]; const float* enb1 = (const float*)d_in[12];
  const float* enW2 = (const float*)d_in[13]; const float* enb2 = (const float*)d_in[14];
  const float* eng = (const float*)d_in[15];  const float* enbt = (const float*)d_in[16];
  const float* eeW0 = (const float*)d_in[17]; const float* eeb0 = (const float*)d_in[18];
  const float* eeW1 = (const float*)d_in[19]; const float* eeb1 = (const float*)d_in[20];
  const float* eeW2 = (const float*)d_in[21]; const float* eeb2 = (const float*)d_in[22];
  const float* eeg = (const float*)d_in[23];  const float* eebt = (const float*)d_in[24];
  const float* mpeW0 = (const float*)d_in[25]; const float* mpeb0 = (const float*)d_in[26];
  const float* mpeW1 = (const float*)d_in[27]; const float* mpeb1 = (const float*)d_in[28];
  const float* mpeW2 = (const float*)d_in[29]; const float* mpeb2 = (const float*)d_in[30];
  const float* mpeg = (const float*)d_in[31];  const float* mpebt = (const float*)d_in[32];
  const float* mpnW0 = (const float*)d_in[33]; const float* mpnb0 = (const float*)d_in[34];
  const float* mpnW1 = (const float*)d_in[35]; const float* mpnb1 = (const float*)d_in[36];
  const float* mpnW2 = (const float*)d_in[37]; const float* mpnb2 = (const float*)d_in[38];
  const float* mpng = (const float*)d_in[39];  const float* mpnbt = (const float*)d_in[40];
  const float* dW0 = (const float*)d_in[41]; const float* db0 = (const float*)d_in[42];
  const float* dW1 = (const float*)d_in[43]; const float* db1 = (const float*)d_in[44];
  const float* dW2 = (const float*)d_in[45]; const float* db2 = (const float*)d_in[46];
  const int* node_type = (const int*)d_in[47];
  const int* senders = (const int*)d_in[48];
  const int* receivers = (const int*)d_in[49];
  (void)n_in; (void)out_size; (void)ws_size; (void)node_type;

  const int N = in_sizes[0] / 3;
  const int E = in_sizes[48];
  const int ntiles_e = (E + 15) / 16;
  const int ntiles_n = N / 16;
  const int STEPS = 15;

  // ---- workspace carve ----
  char* ws = (char*)d_ws;
  size_t off = 0;
  auto carve = [&](size_t bytes) -> void* {
    void* p = ws + off;
    off += (bytes + 255) & ~(size_t)255;
    return p;
  };
  _Float16* en0s = (_Float16*)carve((size_t)32 * 128 * 2);
  _Float16* en1s = (_Float16*)carve((size_t)128 * 128 * 2);
  _Float16* en2s = (_Float16*)carve((size_t)128 * 128 * 2);
  _Float16* ee0s = (_Float16*)carve((size_t)32 * 128 * 2);
  _Float16* ee1s = (_Float16*)carve((size_t)128 * 128 * 2);
  _Float16* ee2s = (_Float16*)carve((size_t)128 * 128 * 2);
  _Float16* me0s = (_Float16*)carve((size_t)STEPS * 384 * 128 * 2);
  _Float16* me1s = (_Float16*)carve((size_t)STEPS * 128 * 128 * 2);
  _Float16* me2s = (_Float16*)carve((size_t)STEPS * 128 * 128 * 2);
  _Float16* mn0s = (_Float16*)carve((size_t)STEPS * 256 * 128 * 2);
  _Float16* mn1s = (_Float16*)carve((size_t)STEPS * 128 * 128 * 2);
  _Float16* mn2s = (_Float16*)carve((size_t)STEPS * 128 * 128 * 2);
  _Float16* d0s  = (_Float16*)carve((size_t)128 * 128 * 2);
  _Float16* d1s  = (_Float16*)carve((size_t)128 * 128 * 2);
  float* node_lat = (float*)carve((size_t)N * L * 4);
  float* edge_lat = (float*)carve((size_t)E * L * 4);
  float* agg      = (float*)carve((size_t)N * L * 4);

  // ---- weight swizzle (fragment-major f16) ----
  auto swz = [&](const float* src, _Float16* dst, int K, int Kpad) {
    int total = Kpad * 128;
    swizzle_w_kernel<<<(total + 255) / 256, 256, 0, stream>>>(src, dst, K, Kpad);
  };
  swz(enW0, en0s, 12, 32); swz(enW1, en1s, 128, 128); swz(enW2, en2s, 128, 128);
  swz(eeW0, ee0s, 7, 32);  swz(eeW1, ee1s, 128, 128); swz(eeW2, ee2s, 128, 128);
  for (int s = 0; s < STEPS; ++s) {
    swz(mpeW0 + (size_t)s * 384 * 128, me0s + (size_t)s * 384 * 128, 384, 384);
    swz(mpeW1 + (size_t)s * 128 * 128, me1s + (size_t)s * 128 * 128, 128, 128);
    swz(mpeW2 + (size_t)s * 128 * 128, me2s + (size_t)s * 128 * 128, 128, 128);
    swz(mpnW0 + (size_t)s * 256 * 128, mn0s + (size_t)s * 256 * 128, 256, 256);
    swz(mpnW1 + (size_t)s * 128 * 128, mn1s + (size_t)s * 128 * 128, 128, 128);
    swz(mpnW2 + (size_t)s * 128 * 128, mn2s + (size_t)s * 128 * 128, 128, 128);
  }
  swz(dW0, d0s, 128, 128); swz(dW1, d1s, 128, 128);

  // ---- encode + zero aggregation buffer ----
  size_t aggN = (size_t)N * L;
  zero_kernel<<<(int)((aggN + 255) / 256), 256, 0, stream>>>(agg, aggN);
  encode_nodes_kernel<<<(ntiles_n + WPB - 1) / WPB, WPB * 32, 0, stream>>>(
      world_pos, prev_world_pos, node_type, node_mean, node_std,
      en0s, en1s, en2s, enb0, enb1, enb2, eng, enbt, node_lat, N);
  encode_edges_kernel<<<(ntiles_e + WPB - 1) / WPB, WPB * 32, 0, stream>>>(
      world_pos, mesh_pos, senders, receivers, edge_mean, edge_std,
      ee0s, ee1s, ee2s, eeb0, eeb1, eeb2, eeg, eebt, edge_lat, E, ntiles_e);

  // ---- 15 message-passing steps ----
  for (int s = 0; s < STEPS; ++s) {
    edge_step_kernel<<<(ntiles_e + WPB - 1) / WPB, WPB * 32, 0, stream>>>(
        senders, receivers, edge_lat, node_lat, agg,
        me0s + (size_t)s * 384 * 128, me1s + (size_t)s * 128 * 128,
        me2s + (size_t)s * 128 * 128,
        mpeb0 + (size_t)s * 128, mpeb1 + (size_t)s * 128, mpeb2 + (size_t)s * 128,
        mpeg + (size_t)s * 128, mpebt + (size_t)s * 128, E, ntiles_e);
    node_step_kernel<<<(ntiles_n + WPB - 1) / WPB, WPB * 32, 0, stream>>>(
        node_lat, agg,
        mn0s + (size_t)s * 256 * 128, mn1s + (size_t)s * 128 * 128,
        mn2s + (size_t)s * 128 * 128,
        mpnb0 + (size_t)s * 128, mpnb1 + (size_t)s * 128, mpnb2 + (size_t)s * 128,
        mpng + (size_t)s * 128, mpnbt + (size_t)s * 128, N);
  }

  // ---- decode + integrate ----
  decode_kernel<<<(ntiles_n + WPB - 1) / WPB, WPB * 32, 0, stream>>>(
      node_lat, d0s, d1s, db0, db1, dW2, db2, out_mean, out_std,
      world_pos, prev_world_pos, (float*)d_out, N);
}
